// ScaledUpOriginalGNN_25812753449152
// MI455X (gfx1250) — compile-verified, matrix operand order checked by reading, not run on hardware
//
#include <hip/hip_runtime.h>
#include <hip/hip_bf16.h>
#include <math.h>

#ifndef __has_builtin
#define __has_builtin(x) 0
#endif
#if __has_builtin(__builtin_amdgcn_tensor_load_to_lds)
#define HAVE_TDM 1
#else
#define HAVE_TDM 0
#endif

// ---------------------------------------------------------------------------
// Problem constants (match reference)
// ---------------------------------------------------------------------------
#define NN 4096            // nodes
#define E0 65536           // original edges
#define ET (E0 + NN)       // edges incl. self loops

typedef __bf16 bf16;
typedef __attribute__((ext_vector_type(16))) __bf16 v16bf;
typedef __attribute__((ext_vector_type(8)))  __bf16 v8bf;
typedef __attribute__((ext_vector_type(8)))  float  v8f;
typedef __attribute__((ext_vector_type(4)))  unsigned int u32x4;
typedef __attribute__((ext_vector_type(8)))  int i32x8;
typedef __attribute__((ext_vector_type(4)))  int i32x4;

// ---------------------------------------------------------------------------
// Float atomic max via signed/unsigned int monotonic mapping
// ---------------------------------------------------------------------------
__device__ __forceinline__ void atomicMaxF(float* addr, float val) {
    if (val >= 0.f) {
        atomicMax((int*)addr, __float_as_int(val));
    } else {
        atomicMin((unsigned int*)addr, (unsigned int)__float_as_int(val));
    }
}

// ---------------------------------------------------------------------------
// WMMA GEMM:  C[M,N] = A[M,K] (bf16) @ Bt[N,K]^T (bf16)  + bias
//
// One workgroup (256 thr = 8 waves) owns one 16-row A panel.  The panel
// (16 x K bf16, <= 32 KB) is staged into LDS ONCE -- via the Tensor Data
// Mover when available (tensor_load_to_lds + s_wait_tensorcnt), else a
// cooperative copy -- then all 8 waves sweep the N tiles reusing it, so the
// A panel is read from HBM exactly once instead of N/16 times.
//
// Fragment layouts per CDNA5 ISA 7.12.2 (16-bit A 16x32 / f32 C 16x16):
//   lane r   (r<16): row r, K = {k..k+7, k+16..k+23}
//   lane r+16      : row r, K = {k+8..k+15, k+24..k+31}
//   C: lane r / r+16 hold col r, acc[v] = row v / v+8.
// K multiple of 32, M and N multiples of 16.
// ---------------------------------------------------------------------------
__global__ void k_gemm_bf16(const bf16* __restrict__ A, const bf16* __restrict__ Bt,
                            const float* __restrict__ bias, float* __restrict__ C,
                            int M, int N, int K)
{
    extern __shared__ bf16 sA[];                    // 16 x K panel
    const int tm   = blockIdx.x;                    // one A panel per block
    const int wv   = threadIdx.x >> 5;
    const int lane = threadIdx.x & 31;
    const int half = lane >> 4;                     // 0 or 1
    const int r    = lane & 15;
    const int tilesN = N >> 4;

#if HAVE_TDM
    if (wv == 0) {
        // ---- Tensor DMA descriptor (D#), 2D tile: 16 rows x K elems of 2B ----
        unsigned long long ga = (unsigned long long)(A + (size_t)tm * 16 * K);
        u32x4 g0;
        g0[0] = 1u;                                  // count=1, user mode
        g0[1] = 0u;                                  // lds_addr = 0 (panel base)
        g0[2] = (unsigned)(ga & 0xffffffffu);        // global_addr[31:0]
        g0[3] = (unsigned)(ga >> 32) | 0x80000000u;  // addr[56:32] | type=2<<30
        i32x8 g1;
        g1[0] = (int)(1u << 16);                     // data_size=1 (2 bytes)
        g1[1] = (int)((unsigned)K << 16);            // tensor_dim0 = K
        g1[2] = (int)(4096u << 16);                  // tensor_dim1 = M rows
        g1[3] = (int)((unsigned)K << 16);            // tile_dim0 = K
        g1[4] = 16;                                  // tile_dim1 = 16 rows
        g1[5] = K;                                   // tensor_dim0_stride = K
        g1[6] = 0;
        g1[7] = 0;
        i32x4 z4 = {0, 0, 0, 0};
        i32x8 z8 = {0, 0, 0, 0, 0, 0, 0, 0};
        // this toolchain's builtin is the 6-arg (clang-23 / therock) form
        __builtin_amdgcn_tensor_load_to_lds(g0, g1, z4, z4, z8, 0);
        __builtin_amdgcn_s_wait_tensorcnt(0);
    }
    __syncthreads();
#else
    // cooperative fallback: copy the panel as dwords
    {
        const unsigned* gsrc = (const unsigned*)(A + (size_t)tm * 16 * K);
        unsigned* ldst = (unsigned*)sA;
        for (int i = threadIdx.x; i < 16 * (K >> 1); i += blockDim.x)
            ldst[i] = gsrc[i];
        __syncthreads();
    }
#endif

    const bf16* larow = sA + (size_t)r * K + half * 8;

    for (int tn = wv; tn < tilesN; tn += 8) {
        const bf16* brow = Bt + (size_t)(tn * 16 + r) * K + half * 8;
        v8f acc = {};
        for (int k = 0; k < K; k += 32) {
            v8bf a0 = *(const v8bf*)(larow + k);        // ds_load_b128
            v8bf a1 = *(const v8bf*)(larow + k + 16);
            v8bf b0 = *(const v8bf*)(brow + k);         // global_load_b128
            v8bf b1 = *(const v8bf*)(brow + k + 16);
            if (k + 32 < K) {
                __builtin_prefetch(brow + k + 32, 0, 1);
            }
            v16bf av, bv;
#pragma unroll
            for (int i = 0; i < 8; ++i) {
                av[i] = a0[i]; av[i + 8] = a1[i];
                bv[i] = b0[i]; bv[i + 8] = b1[i];
            }
            acc = __builtin_amdgcn_wmma_f32_16x16x32_bf16(
                      false, av, false, bv, (short)0, acc, false, false);
        }
        float bb = bias ? bias[tn * 16 + r] : 0.f;
        float* crow = C + (size_t)(tm * 16 + half * 8) * N + tn * 16 + r;
#pragma unroll
        for (int v = 0; v < 8; ++v)
            crow[(size_t)v * N] = acc[v] + bb;
    }
}

// ---------------------------------------------------------------------------
// Elementwise / conversion / utility kernels
// ---------------------------------------------------------------------------
__global__ void k_f2bf(const float* __restrict__ in, bf16* __restrict__ out, int n) {
    int i = blockIdx.x * blockDim.x + threadIdx.x;
    if (i < n) out[i] = (bf16)in[i];
}

// W[K,N] f32 -> Wt[N,K] bf16 (transpose + convert)
__global__ void k_wt_bf(const float* __restrict__ W, bf16* __restrict__ Wt, int K, int N) {
    int i = blockIdx.x * blockDim.x + threadIdx.x;
    if (i >= K * N) return;
    int n = i / K, k = i % K;
    Wt[i] = (bf16)W[(size_t)k * N + n];
}

__global__ void k_fill(float* __restrict__ p, int n, float v) {
    int i = blockIdx.x * blockDim.x + threadIdx.x;
    if (i < n) p[i] = v;
}

__global__ void k_copy(const float* __restrict__ in, float* __restrict__ out, int n) {
    int i = blockIdx.x * blockDim.x + threadIdx.x;
    if (i < n) out[i] = in[i];
}

// ---------------------------------------------------------------------------
// Self-loop edge_attr: mean of incoming edge_attr per node
// ---------------------------------------------------------------------------
__global__ void k_deg_loop(const int* __restrict__ dst, const float* __restrict__ ea,
                           float* __restrict__ deg, float* __restrict__ loop_ea) {
    int e = blockIdx.x * blockDim.x + threadIdx.x;
    if (e >= E0) return;
    int d = dst[e];
    atomicAdd(&deg[d], 1.f);
    atomicAdd(&loop_ea[d * 3 + 0], ea[e * 3 + 0]);
    atomicAdd(&loop_ea[d * 3 + 1], ea[e * 3 + 1]);
    atomicAdd(&loop_ea[d * 3 + 2], ea[e * 3 + 2]);
}

__global__ void k_loop_div(float* __restrict__ loop_ea, const float* __restrict__ deg) {
    int i = blockIdx.x * blockDim.x + threadIdx.x;
    if (i >= NN * 3) return;
    loop_ea[i] /= fmaxf(deg[i / 3], 1.f);
}

// ---------------------------------------------------------------------------
// Attention pieces
// ---------------------------------------------------------------------------
__global__ void k_alpha_nodes(const float* __restrict__ lin,
                              const float* __restrict__ att_src,
                              const float* __restrict__ att_dst,
                              float* __restrict__ asrc, float* __restrict__ adst,
                              int H, int O) {
    int i = blockIdx.x * blockDim.x + threadIdx.x;
    if (i >= NN * H) return;
    int n = i / H, h = i % H;
    const float* row = lin + (size_t)n * H * O + h * O;
    float ss = 0.f, sd = 0.f;
    for (int o = 0; o < O; ++o) {
        float v = row[o];
        ss += v * att_src[h * O + o];
        sd += v * att_dst[h * O + o];
    }
    asrc[i] = ss; adst[i] = sd;
}

// wae[j,h] = sum_o we[j, h*O+o] * att_edge[h,o]   (folds edge-feature GEMM)
__global__ void k_wae(const float* __restrict__ we, const float* __restrict__ att_edge,
                      float* __restrict__ wae, int H, int O) {
    int i = blockIdx.x * blockDim.x + threadIdx.x;
    if (i >= 3 * H) return;
    int j = i / H, h = i % H;
    float s = 0.f;
    for (int o = 0; o < O; ++o)
        s += we[(size_t)j * H * O + h * O + o] * att_edge[h * O + o];
    wae[j * H + h] = s;
}

// raw logits + leaky-relu + segment max (edges E0.. are self loops)
__global__ void k_edge_logits(const int* __restrict__ src, const int* __restrict__ dst,
                              const float* __restrict__ ea, const float* __restrict__ loop_ea,
                              const float* __restrict__ asrc, const float* __restrict__ adst,
                              const float* __restrict__ wae,
                              float* __restrict__ elog, float* __restrict__ segm, int H) {
    int i = blockIdx.x * blockDim.x + threadIdx.x;
    if (i >= ET * H) return;
    int e = i / H, h = i % H;
    int s, d; float e0, e1, e2;
    if (e < E0) {
        s = src[e]; d = dst[e];
        e0 = ea[e * 3]; e1 = ea[e * 3 + 1]; e2 = ea[e * 3 + 2];
    } else {
        s = d = e - E0;
        e0 = loop_ea[s * 3]; e1 = loop_ea[s * 3 + 1]; e2 = loop_ea[s * 3 + 2];
    }
    float a = asrc[s * H + h] + adst[d * H + h]
            + e0 * wae[0 * H + h] + e1 * wae[1 * H + h] + e2 * wae[2 * H + h];
    a = (a > 0.f) ? a : 0.2f * a;
    elog[i] = a;
    atomicMaxF(&segm[d * H + h], a);
}

// exp(logit - segmax) + segment sum of denominators
__global__ void k_edge_exp(const int* __restrict__ dst, float* __restrict__ elog,
                           const float* __restrict__ segm, float* __restrict__ segd, int H) {
    int i = blockIdx.x * blockDim.x + threadIdx.x;
    if (i >= ET * H) return;
    int e = i / H, h = i % H;
    int d = (e < E0) ? dst[e] : (e - E0);
    float ex = __expf(elog[i] - segm[d * H + h]);
    elog[i] = ex;
    atomicAdd(&segd[d * H + h], ex);
}

// weighted scatter-add of messages: agg[d, h*O+o] += lin[s, h*O+o] * alpha
__global__ void k_aggregate(const int* __restrict__ src, const int* __restrict__ dst,
                            const float* __restrict__ lin, const float* __restrict__ elog,
                            const float* __restrict__ segd, float* __restrict__ agg,
                            int H, int O) {
    int HO = H * O;
    int i = blockIdx.x * blockDim.x + threadIdx.x;
    if (i >= ET * HO) return;
    int e = i / HO, c = i % HO, h = c / O;
    int s, d;
    if (e < E0) { s = src[e]; d = dst[e]; } else { s = d = e - E0; }
    float alpha = elog[e * H + h] / fmaxf(segd[d * H + h], 1e-16f);
    atomicAdd(&agg[(size_t)d * HO + c], lin[(size_t)s * HO + c] * alpha);
}

__global__ void k_bias_add(float* __restrict__ x, const float* __restrict__ b, int D) {
    int i = blockIdx.x * blockDim.x + threadIdx.x;
    if (i < NN * D) x[i] += b[i % D];
}

// ---------------------------------------------------------------------------
// BatchNorm (biased var, eps 1e-5) : one block per feature column
// ---------------------------------------------------------------------------
__global__ void k_bn_stats(const float* __restrict__ h, float* __restrict__ stats, int D) {
    int c = blockIdx.x;
    float s = 0.f, sq = 0.f;
    for (int n = threadIdx.x; n < NN; n += blockDim.x) {
        float v = h[(size_t)n * D + c];
        s += v; sq += v * v;
    }
    __shared__ float sh_s[256];
    __shared__ float sh_q[256];
    sh_s[threadIdx.x] = s; sh_q[threadIdx.x] = sq;
    __syncthreads();
    for (int off = 128; off > 0; off >>= 1) {
        if (threadIdx.x < off) {
            sh_s[threadIdx.x] += sh_s[threadIdx.x + off];
            sh_q[threadIdx.x] += sh_q[threadIdx.x + off];
        }
        __syncthreads();
    }
    if (threadIdx.x == 0) {
        float m = sh_s[0] / (float)NN;
        stats[c]     = m;
        stats[D + c] = sh_q[0] / (float)NN - m * m;
    }
}

// y = bn(h)*g+b (+ identity) then ReLU
__global__ void k_bn_apply(const float* __restrict__ h, const float* __restrict__ stats,
                           const float* __restrict__ g, const float* __restrict__ b,
                           const float* __restrict__ idn, float* __restrict__ out, int D) {
    int i = blockIdx.x * blockDim.x + threadIdx.x;
    if (i >= NN * D) return;
    int c = i % D;
    float y = (h[i] - stats[c]) * rsqrtf(stats[D + c] + 1e-5f) * g[c] + b[c];
    if (idn) y += idn[i];
    out[i] = fmaxf(y, 0.f);
}

// ---------------------------------------------------------------------------
// Small dense tail: final 128->5 linear + row log_softmax
// ---------------------------------------------------------------------------
__global__ void k_lin_small(const float* __restrict__ x, const float* __restrict__ w,
                            const float* __restrict__ b, float* __restrict__ out,
                            int K, int C) {
    int i = blockIdx.x * blockDim.x + threadIdx.x;
    if (i >= NN * C) return;
    int n = i / C, c = i % C;
    float s = b[c];
    for (int k = 0; k < K; ++k)
        s += x[(size_t)n * K + k] * w[(size_t)k * C + c];
    out[i] = s;
}

__global__ void k_logsoftmax(const float* __restrict__ in, float* __restrict__ out) {
    int n = blockIdx.x * blockDim.x + threadIdx.x;
    if (n >= NN) return;
    float m = in[n * 5];
    for (int c = 1; c < 5; ++c) m = fmaxf(m, in[n * 5 + c]);
    float s = 0.f;
    for (int c = 0; c < 5; ++c) s += __expf(in[n * 5 + c] - m);
    float ls = logf(s);
    for (int c = 0; c < 5; ++c) out[n * 5 + c] = in[n * 5 + c] - m - ls;
}

// ---------------------------------------------------------------------------
// Host driver
// ---------------------------------------------------------------------------
static inline unsigned gblk(long long n) { return (unsigned)((n + 255) / 256); }

extern "C" void kernel_launch(void* const* d_in, const int* in_sizes, int n_in,
                              void* d_out, int out_size, void* d_ws, size_t ws_size,
                              hipStream_t stream)
{
    (void)in_sizes; (void)n_in; (void)out_size; (void)ws_size;

    // ---- inputs (jax pytree flatten: dicts alphabetical, None dropped) ----
    const float* x     = (const float*)d_in[0];
    const int*   ei    = (const int*)d_in[1];      // [2, E0] : row0=src, row1=dst
    const float* eattr = (const float*)d_in[2];    // [E0, 3]
    const int* src = ei;
    const int* dst = ei + E0;

    int p = 3;
    // params["bns"]: 6 x {b, g}
    const float *bnb[6], *bng[6];
    for (int l = 0; l < 6; ++l) { bnb[l] = (const float*)d_in[p++]; bng[l] = (const float*)d_in[p++]; }
    // params["convs"]: 6 x {att_dst, att_edge, att_src, bias, w, we}
    const float *c_adst[6], *c_aedge[6], *c_asrc[6], *c_bias[6], *c_w[6], *c_we[6];
    for (int l = 0; l < 6; ++l) {
        c_adst[l]  = (const float*)d_in[p++];
        c_aedge[l] = (const float*)d_in[p++];
        c_asrc[l]  = (const float*)d_in[p++];
        c_bias[l]  = (const float*)d_in[p++];
        c_w[l]     = (const float*)d_in[p++];
        c_we[l]    = (const float*)d_in[p++];
    }
    // params["mlp"]: {bn1:{b,g}, bn2:{b,g}, l1:{b,w}, l2:{b,w}, l3:{b,w}}
    const float* mbn1b = (const float*)d_in[p++];
    const float* mbn1g = (const float*)d_in[p++];
    const float* mbn2b = (const float*)d_in[p++];
    const float* mbn2g = (const float*)d_in[p++];
    const float* ml1b  = (const float*)d_in[p++];
    const float* ml1w  = (const float*)d_in[p++];
    const float* ml2b  = (const float*)d_in[p++];
    const float* ml2w  = (const float*)d_in[p++];
    const float* ml3b  = (const float*)d_in[p++];
    const float* ml3w  = (const float*)d_in[p++];
    // params["res"]: layers 0,2,4,5 x {b, w}
    const float *rb[4], *rw[4];
    for (int i = 0; i < 4; ++i) { rb[i] = (const float*)d_in[p++]; rw[i] = (const float*)d_in[p++]; }

    // ---- workspace carve-out (~78 MB) ----
    char* wsp = (char*)d_ws;
    auto carve = [&](size_t bytes) { void* q = wsp; wsp += (bytes + 255) & ~(size_t)255; return q; };
    float* cur     = (float*)carve((size_t)NN * 1024 * 4);
    float* lin     = (float*)carve((size_t)NN * 1024 * 4);
    float* idn     = (float*)carve((size_t)NN * 1024 * 4);
    float* agg     = (float*)carve((size_t)NN * 1024 * 4);
    bf16*  abf     = (bf16*) carve((size_t)NN * 1024 * 2);
    bf16*  wbf     = (bf16*) carve((size_t)1024 * 1024 * 2);
    float* deg     = (float*)carve((size_t)NN * 4);
    float* loop_ea = (float*)carve((size_t)NN * 3 * 4);
    float* asrc    = (float*)carve((size_t)NN * 8 * 4);
    float* adst    = (float*)carve((size_t)NN * 8 * 4);
    float* wae     = (float*)carve(256);
    float* elog    = (float*)carve((size_t)ET * 8 * 4);
    float* segm    = (float*)carve((size_t)NN * 8 * 4);
    float* segd    = (float*)carve((size_t)NN * 8 * 4);
    float* stats   = (float*)carve((size_t)2048 * 4);

    auto gemm = [&](const bf16* A, const bf16* Bt, const float* bias, float* C,
                    int M, int Nc, int K) {
        // one block per 16-row A panel; panel staged in LDS (16*K bf16)
        k_gemm_bf16<<<M / 16, 256, (size_t)16 * K * 2, stream>>>(A, Bt, bias, C, M, Nc, K);
    };

    // ---- layer configs ----
    const int INS_[6] = {32, 512, 512, 1024, 1024, 512};
    const int HO_[6]  = {512, 512, 1024, 1024, 512, 64};
    const int H_[6]   = {8, 8, 8, 8, 8, 1};
    const int O_[6]   = {64, 64, 128, 128, 64, 64};
    const int RIDX[6] = {0, -1, 1, -1, 2, 3};

    // h = x
    (void)hipMemcpyAsync(cur, x, (size_t)NN * 32 * 4, hipMemcpyDeviceToDevice, stream);

    // self-loop attrs + degrees (shared by all layers)
    k_fill<<<gblk(NN), 256, 0, stream>>>(deg, NN, 0.f);
    k_fill<<<gblk(NN * 3), 256, 0, stream>>>(loop_ea, NN * 3, 0.f);
    k_deg_loop<<<gblk(E0), 256, 0, stream>>>(dst, eattr, deg, loop_ea);
    k_loop_div<<<gblk(NN * 3), 256, 0, stream>>>(loop_ea, deg);

    for (int l = 0; l < 6; ++l) {
        const int Din = INS_[l], HO = HO_[l], H = H_[l], O = O_[l];

        // activations -> bf16 once per layer
        k_f2bf<<<gblk((long long)NN * Din), 256, 0, stream>>>(cur, abf, NN * Din);

        // residual path
        if (RIDX[l] >= 0) {
            int ri = RIDX[l];
            k_wt_bf<<<gblk((long long)Din * HO), 256, 0, stream>>>(rw[ri], wbf, Din, HO);
            gemm(abf, wbf, rb[ri], idn, NN, HO, Din);
        } else {
            k_copy<<<gblk((long long)NN * Din), 256, 0, stream>>>(cur, idn, NN * Din);
        }

        // h_lin = h @ W   (WMMA)
        k_wt_bf<<<gblk((long long)Din * HO), 256, 0, stream>>>(c_w[l], wbf, Din, HO);
        gemm(abf, wbf, nullptr, lin, NN, HO, Din);

        // attention scalars
        k_alpha_nodes<<<gblk((long long)NN * H), 256, 0, stream>>>(lin, c_asrc[l], c_adst[l],
                                                                   asrc, adst, H, O);
        k_wae<<<1, 32, 0, stream>>>(c_we[l], c_aedge[l], wae, H, O);

        k_fill<<<gblk(NN * H), 256, 0, stream>>>(segm, NN * H, -3.402823466e38f);
        k_fill<<<gblk(NN * H), 256, 0, stream>>>(segd, NN * H, 0.f);
        k_edge_logits<<<gblk((long long)ET * H), 256, 0, stream>>>(src, dst, eattr, loop_ea,
                                                                   asrc, adst, wae, elog, segm, H);
        k_edge_exp<<<gblk((long long)ET * H), 256, 0, stream>>>(dst, elog, segm, segd, H);

        // aggregate messages
        k_fill<<<gblk((long long)NN * HO), 256, 0, stream>>>(agg, NN * HO, 0.f);
        k_aggregate<<<gblk((long long)ET * HO), 256, 0, stream>>>(src, dst, lin, elog, segd,
                                                                  agg, H, O);
        k_bias_add<<<gblk((long long)NN * HO), 256, 0, stream>>>(agg, c_bias[l], HO);

        // BN + residual + ReLU -> cur
        k_bn_stats<<<HO, 256, 0, stream>>>(agg, stats, HO);
        k_bn_apply<<<gblk((long long)NN * HO), 256, 0, stream>>>(agg, stats, bng[l], bnb[l],
                                                                 idn, cur, HO);
    }

    // ---- MLP head: 64 -> 256 -> 128 -> 5 ----
    k_f2bf<<<gblk((long long)NN * 64), 256, 0, stream>>>(cur, abf, NN * 64);
    k_wt_bf<<<gblk(64 * 256), 256, 0, stream>>>(ml1w, wbf, 64, 256);
    gemm(abf, wbf, ml1b, lin, NN, 256, 64);
    k_bn_stats<<<256, 256, 0, stream>>>(lin, stats, 256);
    k_bn_apply<<<gblk((long long)NN * 256), 256, 0, stream>>>(lin, stats, mbn1g, mbn1b,
                                                              nullptr, cur, 256);

    k_f2bf<<<gblk((long long)NN * 256), 256, 0, stream>>>(cur, abf, NN * 256);
    k_wt_bf<<<gblk(256 * 128), 256, 0, stream>>>(ml2w, wbf, 256, 128);
    gemm(abf, wbf, ml2b, lin, NN, 128, 256);
    k_bn_stats<<<128, 256, 0, stream>>>(lin, stats, 128);
    k_bn_apply<<<gblk((long long)NN * 128), 256, 0, stream>>>(lin, stats, mbn2g, mbn2b,
                                                              nullptr, cur, 128);

    k_lin_small<<<gblk((long long)NN * 5), 256, 0, stream>>>(cur, ml3w, ml3b, lin, 128, 5);
    k_logsoftmax<<<gblk(NN), 256, 0, stream>>>(lin, (float*)d_out);
}
